// NeighbourhoodAttnRegisterBlock_39951785787580
// MI455X (gfx1250) — matrix operand order repaired and checked by reading, not compiled, not run on hardware
//
#include <hip/hip_runtime.h>

typedef __attribute__((ext_vector_type(16))) _Float16 v16h;
typedef __attribute__((ext_vector_type(8)))  _Float16 v8h;
typedef __attribute__((ext_vector_type(8)))  float    v8f;

#define D_MODEL 768
#define THREE_D 2304
#define S_TOK   2304   // 48*48 canvas tokens per batch
#define N_TOK   4608   // 2 * S_TOK
#define R_TOK   16
#define TOK_AUG 4624   // N_TOK + R_TOK (registers appended)
#define CANH    48

__device__ __forceinline__ v16h cat8(v8h lo, v8h hi) {
  return __builtin_shufflevector(lo, hi, 0,1,2,3,4,5,6,7,8,9,10,11,12,13,14,15);
}

// ---------------------------------------------------------------------------
// Generic NT GEMM: C[M,N] = A_f16[M,K] * B_f16[N,K]^T + bias[N]
// One wave per 16(M) x 64(N) tile; K in steps of 32 via v_wmma_f32_16x16x32_f16.
// ---------------------------------------------------------------------------
template<int OUT_F32>
__global__ __launch_bounds__(256) void gemm_nt_wmma(
    const _Float16* __restrict__ A, const _Float16* __restrict__ B,
    const float* __restrict__ bias, void* __restrict__ Cout,
    int M, int N, int K)
{
  const int lane = threadIdx.x & 31;
  const int wave = threadIdx.x >> 5;
  const int gw = blockIdx.x * (blockDim.x >> 5) + wave;
  const int Mt = M >> 4, Nt = N >> 6;
  if (gw >= Mt * Nt) return;                 // wave-uniform: EXEC stays all-ones
  const int mt = gw % Mt, nt = gw / Mt;
  const int m0 = mt << 4, n0 = nt << 6;
  const int half = lane >> 4, lr = lane & 15;

  v8f acc[4] = {};
  const _Float16* arow = A + (size_t)(m0 + lr) * K;
  for (int k0 = 0; k0 < K; k0 += 32) {
    // A fragment: 16x32, row = lr, 16 contiguous halfs per lane (k = 16*half + s)
    v8h a0 = *(const v8h*)(arow + k0 + half * 16);
    v8h a1 = *(const v8h*)(arow + k0 + half * 16 + 8);
    v16h a = cat8(a0, a1);
#pragma unroll
    for (int j = 0; j < 4; ++j) {
      // B fragment: B[k][n] = W[n0+j*16+n][k0+k]; 16 contiguous halfs per lane
      const _Float16* brow = B + (size_t)(n0 + j * 16 + lr) * K + k0 + half * 16;
      v16h bb = cat8(*(const v8h*)brow, *(const v8h*)(brow + 8));
      acc[j] = __builtin_amdgcn_wmma_f32_16x16x32_f16(
          false, a, false, bb, (short)0, acc[j], false, false);
    }
  }
#pragma unroll
  for (int j = 0; j < 4; ++j) {
    const int col = n0 + j * 16 + lr;
    const float bs = bias[col];
#pragma unroll
    for (int i = 0; i < 8; ++i) {
      const int row = m0 + i + half * 8;
      const float v = acc[j][i] + bs;
      if (OUT_F32) ((float*)Cout)[(size_t)row * N + col] = v;
      else         ((_Float16*)Cout)[(size_t)row * N + col] = (_Float16)v;
    }
  }
}

// ---------------------------------------------------------------------------
// Neighbourhood attention. One wave per (batch, head, query-row-strip of 16).
// Candidate keys: 7 rows x 22 cols clamped window (154) + 16 registers = 170,
// padded to 176 = 11 key tiles. QK^T and P*V via WMMA; softmax through LDS.
// ---------------------------------------------------------------------------
__device__ __forceinline__ int key_token(int j, int batch, int sh, int cw0) {
  if (j < 154) {
    const int kh = sh + j / 22;
    const int kw = cw0 + j % 22;
    return batch * S_TOK + kh * CANH + kw;
  }
  int r = j - 154; if (r > 15) r = 15;       // clamp padded keys to a valid row
  return N_TOK + r;                          // register tokens
}

__global__ __launch_bounds__(32) void natten_wmma(
    const _Float16* __restrict__ QKV, _Float16* __restrict__ O)
{
  __shared__ __align__(16) float    sc[16 * 176];   // raw scores (f32)
  __shared__ __align__(16) _Float16 Pm[16 * 192];   // softmax probs (f16, A-layout src)
  __shared__ __align__(16) _Float16 Vt[64 * 192];   // V transposed: [e][key]

  const int wid   = blockIdx.x;
  const int qt    = wid % 3;
  const int qh    = (wid / 3) % 48;
  const int hd    = (wid / 144) % 12;
  const int batch = wid / 1728;
  const int lane  = threadIdx.x;
  const int half  = lane >> 4, lr = lane & 15;
  const int qw0   = qt * 16;
  int sh  = qh - 3;  if (sh < 0) sh = 0;   if (sh > 41) sh = 41;   // window top row
  int cw0 = qw0 - 3; if (cw0 < 0) cw0 = 0; if (cw0 > 26) cw0 = 26; // candidate col base

  // Q fragments (e = 0..63, two 16x32 fragments), row = query lr of the strip
  const int qtok = batch * S_TOK + qh * CANH + qw0 + lr;
  const _Float16* qrow = QKV + (size_t)qtok * THREE_D + hd * 64;
  const v16h qa0 = cat8(*(const v8h*)(qrow + half * 16),
                        *(const v8h*)(qrow + 8 + half * 16));
  const v16h qa1 = cat8(*(const v8h*)(qrow + 32 + half * 16),
                        *(const v8h*)(qrow + 40 + half * 16));

  // ---- scores = Q K^T * scale, masked -------------------------------------
  for (int t = 0; t < 11; ++t) {
    const int jj = t * 16 + lr;
    const int jx = jj > 169 ? 169 : jj;
    const int ktok = key_token(jx, batch, sh, cw0);
    const _Float16* kr = QKV + (size_t)ktok * THREE_D + D_MODEL + hd * 64 + half * 16;
    v8f acc = {};
    v16h b0 = cat8(*(const v8h*)kr,        *(const v8h*)(kr + 8));
    acc = __builtin_amdgcn_wmma_f32_16x16x32_f16(false, qa0, false, b0, (short)0, acc, false, false);
    v16h b1 = cat8(*(const v8h*)(kr + 32), *(const v8h*)(kr + 40));
    acc = __builtin_amdgcn_wmma_f32_16x16x32_f16(false, qa1, false, b1, (short)0, acc, false, false);

    const bool isreg = (jj >= 154) && (jj < 170);
    const bool canv  = jj < 154;
    const int  kw    = cw0 + jj % 22;
#pragma unroll
    for (int i = 0; i < 8; ++i) {
      const int m = i + half * 8;                        // query index in strip
      int sw = qw0 + m - 3; if (sw < 0) sw = 0; if (sw > 41) sw = 41;
      const bool vis = isreg || (canv && kw >= sw && kw < sw + 7);
      sc[m * 176 + jj] = vis ? acc[i] * 0.125f : -1e30f; // scale = 1/sqrt(64)
    }
  }
  __syncthreads();

  // ---- softmax per query row (lanes 0..15, one row each) ------------------
  if (lane < 16) {
    const int m = lane;
    float mx = -1e30f;
    for (int j = 0; j < 176; ++j) mx = fmaxf(mx, sc[m * 176 + j]);
    float sum = 0.f;
    for (int j = 0; j < 176; ++j) sum += __expf(sc[m * 176 + j] - mx);
    const float inv = 1.0f / sum;
    for (int j = 0; j < 176; ++j)
      Pm[m * 192 + j] = (_Float16)(__expf(sc[m * 176 + j] - mx) * inv);
    for (int j = 176; j < 192; ++j) Pm[m * 192 + j] = (_Float16)0.f;
  }

  // ---- gather V transposed into LDS: Vt[e][key] ---------------------------
  {
    const int rsub = lane >> 3;        // 4 key rows per iteration
    const int c    = lane & 7;         // 8-half chunk of the 64-dim V row
    for (int base = 0; base < 170; base += 4) {
      const int j = base + rsub;
      if (j < 170) {
        const int vtok = key_token(j, batch, sh, cw0);
        v8h v = *(const v8h*)(QKV + (size_t)vtok * THREE_D + 2 * D_MODEL + hd * 64 + c * 8);
#pragma unroll
        for (int u = 0; u < 8; ++u) Vt[(c * 8 + u) * 192 + j] = v[u];
      }
    }
    for (int idx = lane; idx < 64 * 22; idx += 32) {     // zero padded key cols
      const int e = idx / 22, j = 170 + idx % 22;
      Vt[e * 192 + j] = (_Float16)0.f;
    }
  }
  __syncthreads();

  // ---- O = P V ------------------------------------------------------------
#pragma unroll
  for (int nt = 0; nt < 4; ++nt) {
    v8f acc = {};
#pragma unroll
    for (int kc = 0; kc < 6; ++kc) {
      const _Float16* pr = &Pm[lr * 192 + kc * 32];
      v16h a = cat8(*(const v8h*)(pr + half * 16), *(const v8h*)(pr + 8 + half * 16));
      const _Float16* vr = &Vt[(nt * 16 + lr) * 192 + kc * 32 + half * 16];
      v16h bb = cat8(*(const v8h*)vr, *(const v8h*)(vr + 8));
      acc = __builtin_amdgcn_wmma_f32_16x16x32_f16(false, a, false, bb, (short)0, acc, false, false);
    }
#pragma unroll
    for (int i = 0; i < 8; ++i) {
      const int m = i + half * 8;
      const int otok = batch * S_TOK + qh * CANH + qw0 + m;
      O[(size_t)otok * D_MODEL + hd * 64 + nt * 16 + lr] = (_Float16)acc[i];
    }
  }
}

// ---------------------------------------------------------------------------
// Elementwise casts
// ---------------------------------------------------------------------------
__global__ void cast_f32_f16(const float* __restrict__ in, _Float16* __restrict__ out, int n) {
  const int i = blockIdx.x * blockDim.x + threadIdx.x;
  if (i < n) out[i] = (_Float16)in[i];
}

__global__ void build_xaug(const float* __restrict__ x, const float* __restrict__ regs,
                           _Float16* __restrict__ xa) {
  const int i = blockIdx.x * blockDim.x + threadIdx.x;
  const int n = TOK_AUG * D_MODEL;
  if (i >= n) return;
  const int row = i / D_MODEL;
  const float v = (row < N_TOK) ? x[i] : regs[i - N_TOK * D_MODEL];
  xa[i] = (_Float16)v;
}

// ---------------------------------------------------------------------------
extern "C" void kernel_launch(void* const* d_in, const int* in_sizes, int n_in,
                              void* d_out, int out_size, void* d_ws, size_t ws_size,
                              hipStream_t stream) {
  const float* x     = (const float*)d_in[0];
  const float* regs  = (const float*)d_in[1];
  const float* qkv_w = (const float*)d_in[2];
  const float* qkv_b = (const float*)d_in[3];
  const float* out_w = (const float*)d_in[4];
  const float* out_b = (const float*)d_in[5];
  float* out = (float*)d_out;

  _Float16* Xaug = (_Float16*)d_ws;                       // [4624, 768]
  _Float16* Wq   = Xaug + (size_t)TOK_AUG * D_MODEL;      // [2304, 768]
  _Float16* Wo   = Wq   + (size_t)THREE_D * D_MODEL;      // [768, 768]
  _Float16* QKV  = Wo   + (size_t)D_MODEL * D_MODEL;      // [4624, 2304]
  _Float16* O16  = QKV  + (size_t)TOK_AUG * THREE_D;      // [4608, 768]

  { int n = TOK_AUG * D_MODEL;
    build_xaug<<<(n + 255) / 256, 256, 0, stream>>>(x, regs, Xaug); }
  { int n = THREE_D * D_MODEL;
    cast_f32_f16<<<(n + 255) / 256, 256, 0, stream>>>(qkv_w, Wq, n); }
  { int n = D_MODEL * D_MODEL;
    cast_f32_f16<<<(n + 255) / 256, 256, 0, stream>>>(out_w, Wo, n); }

  { int waves = (TOK_AUG / 16) * (THREE_D / 64);          // QKV projection (aug)
    gemm_nt_wmma<0><<<(waves + 7) / 8, 256, 0, stream>>>(
        Xaug, Wq, qkv_b, QKV, TOK_AUG, THREE_D, D_MODEL); }

  natten_wmma<<<3456, 32, 0, stream>>>(QKV, O16);         // 2*12*48*3 waves

  { int waves = (N_TOK / 16) * (D_MODEL / 64);            // output projection
    gemm_nt_wmma<1><<<(waves + 7) / 8, 256, 0, stream>>>(
        O16, Wo, out_b, out, N_TOK, D_MODEL, D_MODEL); }
}
